// TransformerLayerForwardAttnToMlp_62998580297923
// MI455X (gfx1250) — matrix operand order repaired and checked by reading, not compiled, not run on hardware
//
#include <hip/hip_runtime.h>
#include <hip/hip_bf16.h>

// ---------------------------------------------------------------------------
// MI455X (gfx1250, wave32) transformer layer:
//   LN1 -> QKV GEMM -> causal flash attention (TDM-staged K tiles)
//   -> out-proj GEMM + residual -> LN2 -> top-1 router
//   -> routed expert FC1(GELU) -> FC2 -> residual
// All matmuls: v_wmma_f32_16x16x32_bf16. GEMMs are double-buffered software
// pipelines: next A tile via Tensor Data Mover (async DMA, TENSORcnt), next
// W tile via vector loads held in registers during the WMMAs, single barrier
// per K-step. MoE uses 64-token M-tiles so expert weights stream L2-friendly.
// ---------------------------------------------------------------------------

typedef __bf16 bf16;
typedef __attribute__((ext_vector_type(16))) __bf16 v16bf;
typedef __attribute__((ext_vector_type(8)))  __bf16 v8bf;
typedef __attribute__((ext_vector_type(2)))  __bf16 v2bf;
typedef __attribute__((ext_vector_type(8)))  float  v8f;
typedef __attribute__((ext_vector_type(4)))  float  v4f;
typedef __attribute__((ext_vector_type(4)))  unsigned int u32x4;
typedef __attribute__((ext_vector_type(8)))  int i32x8;
typedef __attribute__((ext_vector_type(4)))  int i32x4;

#define S_LEN  2048
#define HID    1024
#define NHEAD  16
#define DHEAD  64
#define NEXP   8
#define FFN    4096
#define QKV_N  3072   // NHEAD * 3 * DHEAD

#if __has_builtin(__builtin_amdgcn_tensor_load_to_lds) && \
    __has_builtin(__builtin_amdgcn_s_wait_tensorcnt)
#define HAVE_TDM 1
#else
#define HAVE_TDM 0
#endif

static __device__ __forceinline__ v16bf cat16(v8bf lo, v8bf hi) {
  union { v16bf v; v8bf h[2]; } u;
  u.h[0] = lo; u.h[1] = hi;
  return u.v;
}
static __device__ __forceinline__ v8f zero8() {
  v8f z;
#pragma unroll
  for (int i = 0; i < 8; i++) z[i] = 0.f;
  return z;
}
static __device__ __forceinline__ v8f wmma_bf16(v16bf a, v16bf b, v8f c) {
  return __builtin_amdgcn_wmma_f32_16x16x32_bf16(false, a, false, b,
                                                 (short)0, c, false, false);
}

// 2D tile load Global->LDS via Tensor Data Mover (D# built per ISA 08 §8).
// rows x cols elements of 2-byte data, row stride in elements. No-op when
// the toolchain lacks the builtin (callers keep a vector-load fallback).
static __device__ __forceinline__ void tdm_load_2d_bf16(
    unsigned lds_byte_addr, const void* gptr, unsigned rows, unsigned cols,
    unsigned row_stride_elems) {
#if HAVE_TDM
  unsigned long long ga = (unsigned long long)(uintptr_t)gptr;
  u32x4 g0;
  g0[0] = 1u;                                   // count=1, user descriptor
  g0[1] = lds_byte_addr;                        // lds_addr [63:32]
  g0[2] = (unsigned)(ga & 0xffffffffu);         // global_addr[31:0]
  g0[3] = (unsigned)((ga >> 32) & 0x1ffffffu)   // global_addr[56:32]
          | (2u << 30);                         // type = 2 ("image")
  i32x8 g1;
  g1[0] = (int)(1u << 16);                      // workgroup_mask=0, data_size=1 (2B)
  g1[1] = (int)((cols & 0xffffu) << 16);        // tensor_dim0[15:0] @ bits 63:48
  g1[2] = (int)((cols >> 16) | ((rows & 0xffffu) << 16));  // dim0 hi | dim1 lo
  g1[3] = (int)((rows >> 16) | ((cols & 0xffffu) << 16));  // dim1 hi | tile_dim0
  g1[4] = (int)(rows & 0xffffu);                // tile_dim1 (tile_dim2 = 0)
  g1[5] = (int)row_stride_elems;                // tensor_dim0_stride[31:0]
  g1[6] = 0;                                    // stride hi16 | dim1_stride lo16
  g1[7] = 0;
  i32x4 gz = {0, 0, 0, 0};                      // 2D tensor: groups 2/3 unused
#if defined(__clang_major__) && (__clang_major__ >= 23)
  i32x8 gz8;
#pragma unroll
  for (int i = 0; i < 8; i++) gz8[i] = 0;
  __builtin_amdgcn_tensor_load_to_lds(g0, g1, gz, gz, gz8, 0);
#else
  __builtin_amdgcn_tensor_load_to_lds(g0, g1, gz, gz, 0);
#endif
#endif
}
static __device__ __forceinline__ void tdm_wait0() {
#if HAVE_TDM
  __builtin_amdgcn_s_wait_tensorcnt(0);
#endif
}

// ---------------------------------------------------------------------------
// LayerNorm over H=1024, output bf16. One 256-thread block (8 waves) / token.
// ---------------------------------------------------------------------------
__global__ void ln1_kernel(const float* __restrict__ x,
                           const float* __restrict__ g,
                           const float* __restrict__ b,
                           bf16* __restrict__ out) {
  const int t = blockIdx.x;
  const int tid = threadIdx.x, lane = tid & 31, wave = tid >> 5;
  const float* row = x + (size_t)t * HID;
  __shared__ float red[8], red2[8];
  float v[4], s = 0.f, s2 = 0.f;
#pragma unroll
  for (int i = 0; i < 4; i++) {
    v[i] = row[tid + 256 * i];
    s += v[i];
    s2 += v[i] * v[i];
  }
#pragma unroll
  for (int off = 16; off; off >>= 1) {
    s += __shfl_xor(s, off);
    s2 += __shfl_xor(s2, off);
  }
  if (!lane) { red[wave] = s; red2[wave] = s2; }
  __syncthreads();
  if (tid == 0) {
    float a = 0.f, a2 = 0.f;
    for (int i = 0; i < 8; i++) { a += red[i]; a2 += red2[i]; }
    float m = a / HID;
    red[0] = m;
    red2[0] = rsqrtf(a2 / HID - m * m + 1e-5f);
  }
  __syncthreads();
  const float mean = red[0], rstd = red2[0];
#pragma unroll
  for (int i = 0; i < 4; i++) {
    int c = tid + 256 * i;
    out[(size_t)t * HID + c] = (bf16)((v[i] - mean) * rstd * g[c] + b[c]);
  }
}

// ---------------------------------------------------------------------------
// Generic tiled GEMM: C[64x64 tile] = A_bf16[M,K] @ W_f32->bf16[K,N] + bias.
// BK=32, 256 threads = 8 waves (4 M-subtiles x 2 N-pairs).
// Double-buffered pipeline: A tile k+1 via TDM (dense modes) while WMMAs run
// on tile k; W tile k+1 lives in registers across the compute phase. One
// barrier per K-step.
// MODE 0: QKV      -> bf16 out, q columns (col%192<64) pre-scaled 1/8.
// MODE 1: out-proj -> f32 out = gemm + bias + residual (norm_input).
// MODE 2: MoE FC1  -> gathered tokens, gelu, bf16 out (Hbuf), masked by cnt.
// MODE 3: MoE FC2  -> gathered tokens, f32 out = ni + mprob*(gemm+bias).
// ---------------------------------------------------------------------------
template <int MODE>
__global__ __launch_bounds__(256) void gemm64_kernel(
    const bf16* __restrict__ A, const float* __restrict__ W,
    const float* __restrict__ bias, const float* __restrict__ resid,
    const float* __restrict__ mprob, bf16* __restrict__ outb,
    float* __restrict__ outf, const int* __restrict__ counts,
    const int* __restrict__ lists, int N, int K) {
  __shared__ bf16 lsA[2][64 * 32];   // [m][k], ping-pong
  __shared__ bf16 lsWT[2][64 * 32];  // [n][k], ping-pong
  __shared__ int lsTok[64];
  const int tid = threadIdx.x, lane = tid & 31, wave = tid >> 5;
  const int m_blk = blockIdx.y * 64, n_blk = blockIdx.x * 64;
  int cnt = 0;
  if (MODE >= 2) {
    const int e = blockIdx.z;
    cnt = counts[e];
    if (m_blk >= cnt) return;      // uniform: EXEC stays all-ones for WMMA
    W += (size_t)e * K * N;
    bias += (size_t)e * N;
    if (tid < 64) {
      int idx = m_blk + tid;
      lsTok[tid] = lists[e * S_LEN + (idx < cnt ? idx : cnt - 1)];
    }
    __syncthreads();
  }
  constexpr bool kTdmA = (MODE <= 1) && (HAVE_TDM != 0);
  const int m_off = (wave & 3) * 16, n_off = (wave >> 2) * 32;
  const int half = lane >> 4, l16 = lane & 15;
  const int arow = tid >> 2, acol = (tid & 3) * 8;  // A: 64x32 bf16, 8/thread
  const int wrp = tid >> 4, wcg = tid & 15;         // W: 2 k-rows x 4 cols
  const size_t aRow = (MODE >= 2) ? (size_t)lsTok[arow] : (size_t)(m_blk + arow);
  v8f acc0 = zero8(), acc1 = zero8();
  const int nIter = K / 32;

  // ---- prologue: stage tile 0 into buffer 0 ----
  if (kTdmA) {
    if (wave == 0)
      tdm_load_2d_bf16((unsigned)(uintptr_t)(void*)lsA[0],
                       A + (size_t)m_blk * K, 64, 32, K);
  } else {
    *(v8bf*)(lsA[0] + arow * 32 + acol) = *(const v8bf*)(A + aRow * K + acol);
  }
  {
    const float* wp0 = W + (size_t)(2 * wrp) * N + n_blk + wcg * 4;
    v4f wa = *(const v4f*)wp0;
    v4f wb = *(const v4f*)(wp0 + N);
#pragma unroll
    for (int j = 0; j < 4; j++) {
      v2bf p; p[0] = (bf16)wa[j]; p[1] = (bf16)wb[j];
      *(v2bf*)(lsWT[0] + (wcg * 4 + j) * 32 + 2 * wrp) = p;
    }
  }
  if (kTdmA && wave == 0) tdm_wait0();
  __syncthreads();

  // ---- pipelined main loop: one barrier per K-step ----
  for (int it = 0; it < nIter; ++it) {
    const int k0 = it * 32;
    const int cur = it & 1, nxt = cur ^ 1;
    const bool hasNext = (it + 1 < nIter);

    // kick off tile k+1: A via TDM (dense) or vector load; W to registers
    v8bf avn;
    v4f wan, wbn;
    if (hasNext) {
      if (kTdmA) {
        if (wave == 0)
          tdm_load_2d_bf16((unsigned)(uintptr_t)(void*)lsA[nxt],
                           A + (size_t)m_blk * K + k0 + 32, 64, 32, K);
      } else {
        avn = *(const v8bf*)(A + aRow * K + k0 + 32 + acol);
      }
      const float* wp0 = W + (size_t)(k0 + 32 + 2 * wrp) * N + n_blk + wcg * 4;
      wan = *(const v4f*)wp0;
      wbn = *(const v4f*)(wp0 + N);
      if (k0 + 64 < K)
        __builtin_prefetch(wp0 + (size_t)32 * N, 0, 1);
    }

    // compute on buffer `cur` (global loads above are still in flight)
    const int ar = m_off + l16;
    v16bf af = cat16(*(const v8bf*)(lsA[cur] + ar * 32 + 8 * half),
                     *(const v8bf*)(lsA[cur] + ar * 32 + 16 + 8 * half));
    const int bn0 = n_off + l16;
    v16bf bf0 = cat16(*(const v8bf*)(lsWT[cur] + bn0 * 32 + 8 * half),
                      *(const v8bf*)(lsWT[cur] + bn0 * 32 + 16 + 8 * half));
    const int bn1 = bn0 + 16;
    v16bf bf1 = cat16(*(const v8bf*)(lsWT[cur] + bn1 * 32 + 8 * half),
                      *(const v8bf*)(lsWT[cur] + bn1 * 32 + 16 + 8 * half));
    acc0 = wmma_bf16(af, bf0, acc0);
    acc1 = wmma_bf16(af, bf1, acc1);

    // publish tile k+1 into buffer `nxt`, then one barrier
    if (hasNext) {
      if (!kTdmA) *(v8bf*)(lsA[nxt] + arow * 32 + acol) = avn;
#pragma unroll
      for (int j = 0; j < 4; j++) {
        v2bf p; p[0] = (bf16)wan[j]; p[1] = (bf16)wbn[j];
        *(v2bf*)(lsWT[nxt] + (wcg * 4 + j) * 32 + 2 * wrp) = p;
      }
      if (kTdmA && wave == 0) tdm_wait0();
      __syncthreads();
    }
  }

#pragma unroll
  for (int r = 0; r < 8; r++) {
    const int rl = m_off + half * 8 + r;
    const int c0 = n_blk + n_off + l16;
    const int c1 = c0 + 16;
    float v0 = acc0[r] + bias[c0];
    float v1 = acc1[r] + bias[c1];
    if (MODE == 0) {
      const size_t row = m_blk + rl;
      if ((c0 % 192) < 64) v0 *= 0.125f;   // fold 1/sqrt(DH) into q
      if ((c1 % 192) < 64) v1 *= 0.125f;
      outb[row * N + c0] = (bf16)v0;
      outb[row * N + c1] = (bf16)v1;
    } else if (MODE == 1) {
      const size_t row = m_blk + rl;
      outf[row * N + c0] = v0 + resid[row * N + c0];
      outf[row * N + c1] = v1 + resid[row * N + c1];
    } else if (MODE == 2) {
      if (m_blk + rl < cnt) {
        const size_t row = lsTok[rl];
        outb[row * N + c0] = (bf16)(0.5f * v0 * (1.f + erff(v0 * 0.70710678f)));
        outb[row * N + c1] = (bf16)(0.5f * v1 * (1.f + erff(v1 * 0.70710678f)));
      }
    } else {
      if (m_blk + rl < cnt) {
        const size_t row = lsTok[rl];
        const float p = mprob[row];
        outf[row * N + c0] = resid[row * N + c0] + p * v0;
        outf[row * N + c1] = resid[row * N + c1] + p * v1;
      }
    }
  }
}

// ---------------------------------------------------------------------------
// Causal flash attention, one block = (64 queries, 1 head). 128 thr = 4 waves,
// each wave owns 16 query rows x full DH=64 (4 f32 accum frags).
// K tile staged by the Tensor Data Mover (wave 0 issues, s_wait_tensorcnt),
// overlapped with the V-transpose vector loads.
// ---------------------------------------------------------------------------
__global__ __launch_bounds__(128, 1) void attn_kernel(
    const bf16* __restrict__ qkv, bf16* __restrict__ ctx) {
  __shared__ bf16 lsQ[64 * 64];
  __shared__ bf16 lsK[64 * 64];     // [key][dh]  == B layout for Q@K^T
  __shared__ bf16 lsVT[64 * 64];    // [dh][key]  == B layout for P@V
  __shared__ bf16 lsP[4][16 * 64];  // per-wave probs tile [row][key]
  const int tid = threadIdx.x, lane = tid & 31, wave = tid >> 5;
  const int qb = blockIdx.x, head = blockIdx.y;
  const int half = lane >> 4, l16 = lane & 15;
  const int hq = head * 192, hk = hq + 64, hv = hq + 128;

#pragma unroll
  for (int i = 0; i < 4; i++) {
    int vidx = tid * 4 + i;               // 512 x v8bf covers 64x64
    int row = vidx >> 3, col8 = (vidx & 7) * 8;
    *(v8bf*)(lsQ + row * 64 + col8) =
        *(const v8bf*)(qkv + (size_t)(qb * 64 + row) * QKV_N + hq + col8);
  }
  __syncthreads();

  v16bf qf[2];
  {
    const int ar = wave * 16 + l16;
    qf[0] = cat16(*(const v8bf*)(lsQ + ar * 64 + 8 * half),
                  *(const v8bf*)(lsQ + ar * 64 + 16 + 8 * half));
    qf[1] = cat16(*(const v8bf*)(lsQ + ar * 64 + 32 + 8 * half),
                  *(const v8bf*)(lsQ + ar * 64 + 48 + 8 * half));
  }

  float m_i[8], l_i[8];
  v8f oacc[4];
#pragma unroll
  for (int r = 0; r < 8; r++) { m_i[r] = -1e30f; l_i[r] = 0.f; }
#pragma unroll
  for (int nt = 0; nt < 4; nt++) oacc[nt] = zero8();

  const int vrp = tid >> 2, vcg = (tid & 3) * 16;  // V: 2 rows x 16 cols

  for (int kb = 0; kb <= qb; kb++) {
    __syncthreads();  // previous iteration's K/V readers done
    if (HAVE_TDM) {
      if (wave == 0)
        tdm_load_2d_bf16((unsigned)(uintptr_t)(void*)lsK,
                         qkv + (size_t)(kb * 64) * QKV_N + hk, 64, 64, QKV_N);
    } else {
#pragma unroll
      for (int i = 0; i < 4; i++) {
        int vidx = tid * 4 + i;
        int row = vidx >> 3, col8 = (vidx & 7) * 8;
        *(v8bf*)(lsK + row * 64 + col8) =
            *(const v8bf*)(qkv + (size_t)(kb * 64 + row) * QKV_N + hk + col8);
      }
    }
    // V tile: transpose into lsVT with packed pair stores
    {
      const bf16* vp0 = qkv + (size_t)(kb * 64 + 2 * vrp) * QKV_N + hv + vcg;
      const bf16* vp1 = vp0 + QKV_N;
      v8bf a0 = *(const v8bf*)vp0;
      v8bf a1 = *(const v8bf*)(vp0 + 8);
      v8bf b0 = *(const v8bf*)vp1;
      v8bf b1 = *(const v8bf*)(vp1 + 8);
#pragma unroll
      for (int j = 0; j < 8; j++) {
        v2bf p; p[0] = a0[j]; p[1] = b0[j];
        *(v2bf*)(lsVT + (vcg + j) * 64 + 2 * vrp) = p;
      }
#pragma unroll
      for (int j = 0; j < 8; j++) {
        v2bf p; p[0] = a1[j]; p[1] = b1[j];
        *(v2bf*)(lsVT + (vcg + 8 + j) * 64 + 2 * vrp) = p;
      }
    }
    if (HAVE_TDM && wave == 0) tdm_wait0();
    __syncthreads();

    // scores S = Q @ K^T  (q pre-scaled by 1/8)
    float st[4][8];
#pragma unroll
    for (int nt = 0; nt < 4; nt++) {
      v8f sacc = zero8();
#pragma unroll
      for (int kc = 0; kc < 2; kc++) {
        const int bk = nt * 16 + l16;
        v16bf bfr = cat16(*(const v8bf*)(lsK + bk * 64 + kc * 32 + 8 * half),
                          *(const v8bf*)(lsK + bk * 64 + kc * 32 + 16 + 8 * half));
        sacc = wmma_bf16(qf[kc], bfr, sacc);
      }
#pragma unroll
      for (int r = 0; r < 8; r++) {
        float sv = sacc[r];
        if (kb == qb) {
          const int qrow = wave * 16 + r + 8 * half;
          const int krow = nt * 16 + l16;
          if (krow > qrow) sv = -1e30f;   // causal mask
        }
        st[nt][r] = sv;
      }
    }

    // online softmax update (each row lives on 16 lanes of a half-wave)
#pragma unroll
    for (int r = 0; r < 8; r++) {
      float pm = fmaxf(fmaxf(st[0][r], st[1][r]), fmaxf(st[2][r], st[3][r]));
#pragma unroll
      for (int off = 8; off; off >>= 1) pm = fmaxf(pm, __shfl_xor(pm, off));
      const float m_new = fmaxf(m_i[r], pm);
      const float scale = __expf(m_i[r] - m_new);
      float rsum = 0.f;
#pragma unroll
      for (int nt = 0; nt < 4; nt++) {
        float p = __expf(st[nt][r] - m_new);
        st[nt][r] = p;
        rsum += p;
      }
#pragma unroll
      for (int off = 8; off; off >>= 1) rsum += __shfl_xor(rsum, off);
      l_i[r] = l_i[r] * scale + rsum;
      m_i[r] = m_new;
#pragma unroll
      for (int nt = 0; nt < 4; nt++) oacc[nt][r] *= scale;
    }

    // D-layout -> A-layout transpose of P via per-wave LDS slab
#pragma unroll
    for (int nt = 0; nt < 4; nt++)
#pragma unroll
      for (int r = 0; r < 8; r++)
        lsP[wave][(r + 8 * half) * 64 + nt * 16 + l16] = (bf16)st[nt][r];

    v16bf pf[2];
    pf[0] = cat16(*(const v8bf*)(lsP[wave] + l16 * 64 + 8 * half),
                  *(const v8bf*)(lsP[wave] + l16 * 64 + 16 + 8 * half));
    pf[1] = cat16(*(const v8bf*)(lsP[wave] + l16 * 64 + 32 + 8 * half),
                  *(const v8bf*)(lsP[wave] + l16 * 64 + 48 + 8 * half));
#pragma unroll
    for (int nt = 0; nt < 4; nt++) {
#pragma unroll
      for (int kc = 0; kc < 2; kc++) {
        const int bn = nt * 16 + l16;
        v16bf bv = cat16(*(const v8bf*)(lsVT + bn * 64 + kc * 32 + 8 * half),
                         *(const v8bf*)(lsVT + bn * 64 + kc * 32 + 16 + 8 * half));
        oacc[nt] = wmma_bf16(pf[kc], bv, oacc[nt]);
      }
    }
  }

#pragma unroll
  for (int nt = 0; nt < 4; nt++)
#pragma unroll
    for (int r = 0; r < 8; r++) {
      const int row = qb * 64 + wave * 16 + r + 8 * half;
      const int col = head * 64 + nt * 16 + l16;
      ctx[(size_t)row * HID + col] = (bf16)(oacc[nt][r] / l_i[r]);
    }
}

// ---------------------------------------------------------------------------
// LN2 + top-1 router + atomic token compaction per expert.
// ---------------------------------------------------------------------------
__global__ void ln2_router_kernel(const float* __restrict__ ni,
                                  const float* __restrict__ g,
                                  const float* __restrict__ b,
                                  const float* __restrict__ rw,
                                  const float* __restrict__ rb,
                                  bf16* __restrict__ ybf,
                                  float* __restrict__ maxprob,
                                  int* __restrict__ counts,
                                  int* __restrict__ lists) {
  __shared__ float ylds[HID];
  __shared__ float red[8], red2[8];
  __shared__ float logits[NEXP];
  const int t = blockIdx.x, tid = threadIdx.x, lane = tid & 31, wave = tid >> 5;
  const float* row = ni + (size_t)t * HID;
  float v[4], s = 0.f, s2 = 0.f;
#pragma unroll
  for (int i = 0; i < 4; i++) {
    v[i] = row[tid + 256 * i];
    s += v[i];
    s2 += v[i] * v[i];
  }
#pragma unroll
  for (int off = 16; off; off >>= 1) {
    s += __shfl_xor(s, off);
    s2 += __shfl_xor(s2, off);
  }
  if (!lane) { red[wave] = s; red2[wave] = s2; }
  __syncthreads();
  if (tid == 0) {
    float a = 0.f, a2 = 0.f;
    for (int i = 0; i < 8; i++) { a += red[i]; a2 += red2[i]; }
    float m = a / HID;
    red[0] = m;
    red2[0] = rsqrtf(a2 / HID - m * m + 1e-5f);
  }
  __syncthreads();
  const float mean = red[0], rstd = red2[0];
#pragma unroll
  for (int i = 0; i < 4; i++) {
    int c = tid + 256 * i;
    float y = (v[i] - mean) * rstd * g[c] + b[c];
    ylds[c] = y;
    ybf[(size_t)t * HID + c] = (bf16)y;
  }
  __syncthreads();
  {   // one wave per expert: logit = y . router_w[:, e]
    float acc = 0.f;
    for (int i = lane; i < HID; i += 32) acc += ylds[i] * rw[i * NEXP + wave];
#pragma unroll
    for (int off = 16; off; off >>= 1) acc += __shfl_xor(acc, off);
    if (!lane) logits[wave] = acc + rb[wave];
  }
  __syncthreads();
  if (tid == 0) {
    float best = logits[0];
    int bi = 0;
    for (int e = 1; e < NEXP; e++)
      if (logits[e] > best) { best = logits[e]; bi = e; }
    float ssum = 0.f;
    for (int e = 0; e < NEXP; e++) ssum += __expf(logits[e] - best);
    maxprob[t] = 1.f / ssum;            // softmax prob of the argmax expert
    int pos = atomicAdd(&counts[bi], 1);
    lists[bi * S_LEN + pos] = t;        // list order irrelevant to values
  }
}

__global__ void zero_counts_kernel(int* __restrict__ counts) {
  if (threadIdx.x < NEXP) counts[threadIdx.x] = 0;
}

// ---------------------------------------------------------------------------
extern "C" void kernel_launch(void* const* d_in, const int* in_sizes, int n_in,
                              void* d_out, int out_size, void* d_ws,
                              size_t ws_size, hipStream_t stream) {
  const float* hidden = (const float*)d_in[0];
  const float* ln1_g  = (const float*)d_in[1];
  const float* ln1_b  = (const float*)d_in[2];
  const float* w_qkv  = (const float*)d_in[3];
  const float* b_qkv  = (const float*)d_in[4];
  const float* w_o    = (const float*)d_in[5];
  const float* b_o    = (const float*)d_in[6];
  const float* ln2_g  = (const float*)d_in[7];
  const float* ln2_b  = (const float*)d_in[8];
  const float* r_w    = (const float*)d_in[9];
  const float* r_b    = (const float*)d_in[10];
  const float* w1     = (const float*)d_in[11];
  const float* b1     = (const float*)d_in[12];
  const float* w2     = (const float*)d_in[13];
  const float* b2     = (const float*)d_in[14];
  // d_in[15] = attention_mask: standard causal triu(k=1); applied analytically.

  char* ws = (char*)d_ws;
  bf16*  x_bf   = (bf16*)(ws + (0ull  << 20));  //  4 MiB  [S,H] bf16
  bf16*  qkv_bf = (bf16*)(ws + (4ull  << 20));  // 12 MiB  [S,3H] bf16
  bf16*  ctx_bf = (bf16*)(ws + (16ull << 20));  //  4 MiB  [S,H] bf16
  float* ni     = (float*)(ws + (20ull << 20)); //  8 MiB  norm_input f32
  bf16*  y_bf   = (bf16*)(ws + (28ull << 20));  //  4 MiB  [S,H] bf16
  float* mprob  = (float*)(ws + (32ull << 20)); //  8 KiB
  int*   counts = (int*)  (ws + (32ull << 20) + 16384);
  int*   lists  = (int*)  (ws + (32ull << 20) + 16384 + 256); // 64 KiB
  bf16*  hbuf   = (bf16*)(ws + (33ull << 20));  // 16 MiB  [S,FFN] bf16
  float* out    = (float*)d_out;

  zero_counts_kernel<<<dim3(1), dim3(32), 0, stream>>>(counts);

  ln1_kernel<<<dim3(S_LEN), dim3(256), 0, stream>>>(hidden, ln1_g, ln1_b, x_bf);

  gemm64_kernel<0><<<dim3(QKV_N / 64, S_LEN / 64), dim3(256), 0, stream>>>(
      x_bf, w_qkv, b_qkv, nullptr, nullptr, qkv_bf, nullptr, nullptr, nullptr,
      QKV_N, HID);

  attn_kernel<<<dim3(S_LEN / 64, NHEAD), dim3(128), 0, stream>>>(qkv_bf, ctx_bf);

  gemm64_kernel<1><<<dim3(HID / 64, S_LEN / 64), dim3(256), 0, stream>>>(
      ctx_bf, w_o, b_o, hidden, nullptr, nullptr, ni, nullptr, nullptr,
      HID, HID);

  ln2_router_kernel<<<dim3(S_LEN), dim3(256), 0, stream>>>(
      ni, ln2_g, ln2_b, r_w, r_b, y_bf, mprob, counts, lists);

  gemm64_kernel<2><<<dim3(FFN / 64, S_LEN / 64, NEXP), dim3(256), 0, stream>>>(
      y_bf, w1, b1, nullptr, nullptr, hbuf, nullptr, counts, lists, FFN, HID);

  gemm64_kernel<3><<<dim3(HID / 64, S_LEN / 64, NEXP), dim3(256), 0, stream>>>(
      hbuf, w2, b2, ni, mprob, nullptr, out, counts, lists, HID, FFN);
}
